// RNN_18373870092537
// MI455X (gfx1250) — matrix operand order repaired and checked by reading, not compiled, not run on hardware
//
#include <hip/hip_runtime.h>

typedef __attribute__((ext_vector_type(16))) _Float16 v16h;
typedef __attribute__((ext_vector_type(8)))  _Float16 v8h;
typedef __attribute__((ext_vector_type(8)))  float    v8f;

#define T_STEPS  1000
#define BATCH    256
#define FIN      39
#define HDIM     100
#define NCLS     20
#define G3       300      // 3*H gate rows
#define GPAD     312      // padded gate rows (chunk 6 reaches 2*H+111 = 311)
#define KIN      64       // K for input gemm, 39 padded to 64 (2 wmma K-steps)
#define KH       128      // K for hidden gemm, 100 padded to 128 (4 wmma K-steps)
#define HPAD     112      // 7 chunks * 16
#define ROWS     16       // batch rows per workgroup
#define NTHREADS 224      // 7 waves (wave32); wave w owns H-chunk w
#define NWG      (BATCH / ROWS)

static __device__ inline v8f init8(float v) {
    v8f a;
#pragma unroll
    for (int i = 0; i < 8; ++i) a[i] = v;
    return a;
}

// A operand (contiguous K order chosen here, B loads use the same): lane half j -> K = (lane<16?0:16) + j
// => two contiguous 8-half (16B) loads per 32-wide K step.
static __device__ inline v16h loadA(const _Float16* p, int kb, int off) {
    const v8h lo = *(const v8h*)(p + kb + off);
    const v8h hi = *(const v8h*)(p + kb + 8 + off);
    v16h a;
#pragma unroll
    for (int i = 0; i < 8; ++i) { a[i] = lo[i]; a[i + 8] = hi[i]; }
    return a;
}

static __device__ inline v8f wmma_f16(v16h a, v16h b, v8f c) {
    return __builtin_amdgcn_wmma_f32_16x16x32_f16(false, a, false, b, (short)0, c, false, false);
}

static __device__ inline float sigmoidf_(float x) { return 1.0f / (1.0f + __expf(-x)); }

__global__ __launch_bounds__(NTHREADS) void gru_fused_kernel(
    const float* __restrict__ mfcc0, const float* __restrict__ mfcc1,
    const float* __restrict__ mfcc2, const float* __restrict__ len0,
    const float* __restrict__ W_ih,  const float* __restrict__ W_hh,
    const float* __restrict__ b_ih,  const float* __restrict__ b_hh,
    const float* __restrict__ W_out, const float* __restrict__ b_out,
    float* __restrict__ out)
{
    __shared__ __align__(32) _Float16 sWih[GPAD][KIN];   // f16 W_ih, K padded w/ zeros
    __shared__ __align__(32) _Float16 sWhh[GPAD][KH];    // f16 W_hh, K padded w/ zeros
    __shared__ __align__(32) _Float16 sX[ROWS][KIN];     // x_t tile (A operand)
    __shared__ __align__(32) _Float16 sH16[ROWS][KH];    // h as f16 (A operand), pads stay 0
    __shared__ float sH32[ROWS][HPAD];                   // h as f32 (blend + precision)
    __shared__ float sFeat[ROWS][2 * HDIM];              // [avg | max] features

    const int tid    = threadIdx.x;
    const int wave   = tid >> 5;
    const int lane   = tid & 31;
    const int wgbase = blockIdx.x * ROWS;

    // ---------------- one-time LDS init (f32 -> f16 weight conversion) ----------------
    for (int e = tid; e < GPAD * KIN; e += NTHREADS) {
        int r = e / KIN, k = e % KIN;
        sWih[r][k] = (r < G3 && k < FIN) ? (_Float16)W_ih[r * FIN + k] : (_Float16)0.0f;
    }
    for (int e = tid; e < GPAD * KH; e += NTHREADS) {
        int r = e / KH, k = e % KH;
        sWhh[r][k] = (r < G3 && k < HDIM) ? (_Float16)W_hh[r * HDIM + k] : (_Float16)0.0f;
    }
    for (int e = tid; e < ROWS * KIN; e += NTHREADS)  (&sX[0][0])[e]   = (_Float16)0.0f;
    for (int e = tid; e < ROWS * KH; e += NTHREADS)   (&sH16[0][0])[e] = (_Float16)0.0f;
    for (int e = tid; e < ROWS * HPAD; e += NTHREADS) (&sH32[0][0])[e] = 0.0f;

    // ---------------- per-lane constants ----------------
    const int  ncol    = lane & 15;                 // output column within 16-wide tile
    const int  rowbase = (lane < 16) ? 0 : 8;       // C/D layout: lanes 16-31 hold M=8..15
    const int  col     = wave * 16 + ncol;          // hidden-dim column 0..111
    const bool colOK   = (col < HDIM);

    const float bir = colOK ? b_ih[col]            : 0.0f;
    const float biz = colOK ? b_ih[HDIM + col]     : 0.0f;
    const float bin = colOK ? b_ih[2 * HDIM + col] : 0.0f;
    const float bhr = colOK ? b_hh[col]            : 0.0f;
    const float bhz = colOK ? b_hh[HDIM + col]     : 0.0f;
    const float bhn = colOK ? b_hh[2 * HDIM + col] : 0.0f;

    float invlen[8], sumv[8], maxv[8];
#pragma unroll
    for (int r = 0; r < 8; ++r) {
        invlen[r] = 1.0f / len0[wgbase + rowbase + r];
        sumv[r]   = 0.0f;
        maxv[r]   = -3.0e38f;
    }

    const int am   = lane & 15;                 // A row (batch row) for this lane
    const int aoff = (lane < 16) ? 0 : 16;      // A K-offset per lane half
    const int boff = (lane < 16) ? 0 : 16;      // B K-offset per lane half
    const int g_r  = 0 * HDIM + col;            // W row indices (fit in GPAD=312)
    const int g_z  = 1 * HDIM + col;
    const int g_n  = 2 * HDIM + col;

    // ---------------- recurrence over T ----------------
    for (int t = 0; t < T_STEPS; ++t) {
        __syncthreads();  // h(t-1) writes visible; x(t-1) consumers done

        // stage x_t tile (concat of 3 mfcc streams) as f16 into LDS
        for (int e = tid; e < ROWS * FIN; e += NTHREADS) {
            int r = e / FIN, f = e % FIN;
            int b = wgbase + r;
            float v;
            if (f < 13)      v = mfcc0[(t * BATCH + b) * 13 + f];
            else if (f < 26) v = mfcc1[(t * BATCH + b) * 13 + (f - 13)];
            else             v = mfcc2[(t * BATCH + b) * 13 + (f - 26)];
            sX[r][f] = (_Float16)v;
        }
        if (t + 1 < T_STEPS) {  // gfx1250 global_prefetch_b8 for next step's tile
            __builtin_prefetch(&mfcc0[((t + 1) * BATCH + wgbase) * 13], 0, 0);
            __builtin_prefetch(&mfcc1[((t + 1) * BATCH + wgbase) * 13], 0, 0);
            __builtin_prefetch(&mfcc2[((t + 1) * BATCH + wgbase) * 13], 0, 0);
        }
        __syncthreads();

        // accumulators seeded with gate biases (bias is constant per output column
        // = per lane, per the C/D VGPR layout)
        v8f ir = init8(bir), iz = init8(biz), in_ = init8(bin);
        v8f hr = init8(bhr), hz = init8(bhz), hn  = init8(bhn);

        // input projection: [16 x 64] x [64 x 16] per gate (K steps 0,32)
        {
            v16h a0 = loadA(&sX[am][0], 0, aoff);
            v16h a1 = loadA(&sX[am][0], 32, aoff);
            v16h br0 = *(const v16h*)&sWih[g_r][0 + boff];
            v16h br1 = *(const v16h*)&sWih[g_r][32 + boff];
            v16h bz0 = *(const v16h*)&sWih[g_z][0 + boff];
            v16h bz1 = *(const v16h*)&sWih[g_z][32 + boff];
            v16h bn0 = *(const v16h*)&sWih[g_n][0 + boff];
            v16h bn1 = *(const v16h*)&sWih[g_n][32 + boff];
            ir = wmma_f16(a0, br0, ir); ir = wmma_f16(a1, br1, ir);
            iz = wmma_f16(a0, bz0, iz); iz = wmma_f16(a1, bz1, iz);
            in_ = wmma_f16(a0, bn0, in_); in_ = wmma_f16(a1, bn1, in_);
        }

        // hidden projection: [16 x 128] x [128 x 16] per gate (K steps 0,32,64,96)
#pragma unroll
        for (int kb = 0; kb < KH; kb += 32) {
            v16h a  = loadA(&sH16[am][0], kb, aoff);
            v16h br = *(const v16h*)&sWhh[g_r][kb + boff];
            v16h bz = *(const v16h*)&sWhh[g_z][kb + boff];
            v16h bn = *(const v16h*)&sWhh[g_n][kb + boff];
            hr = wmma_f16(a, br, hr);
            hz = wmma_f16(a, bz, hz);
            hn = wmma_f16(a, bn, hn);
        }

        __syncthreads();  // all waves finished reading sH16 before anyone rewrites it

        // GRU gate math — elementwise, fully in-register (matching lane/element maps)
#pragma unroll
        for (int r = 0; r < 8; ++r) {
            int   m    = rowbase + r;
            float rg   = sigmoidf_(ir[r] + hr[r]);
            float zg   = sigmoidf_(iz[r] + hz[r]);
            float ng   = tanhf(in_[r] + rg * hn[r]);
            float hold = sH32[m][col];
            float hnew = (1.0f - zg) * ng + zg * hold;
            if (colOK) {
                sH32[m][col] = hnew;
                sH16[m][col] = (_Float16)hnew;
                sumv[r] += hnew;
                maxv[r]  = fmaxf(maxv[r], hnew);
            }
        }
    }

    // ---------------- pooling + classifier ----------------
    __syncthreads();
    if (colOK) {
#pragma unroll
        for (int r = 0; r < 8; ++r) {
            int m = rowbase + r;
            sFeat[m][col]        = sumv[r] * invlen[r];  // avg pool
            sFeat[m][HDIM + col] = maxv[r];              // max pool
        }
    }
    __syncthreads();

    for (int o = tid; o < ROWS * NCLS; o += NTHREADS) {
        int m = o / NCLS, cls = o % NCLS;
        float acc = b_out[cls];
        for (int k = 0; k < 2 * HDIM; ++k)
            acc += sFeat[m][k] * W_out[cls * 2 * HDIM + k];
        out[(wgbase + m) * NCLS + cls] = acc;
    }
}

extern "C" void kernel_launch(void* const* d_in, const int* in_sizes, int n_in,
                              void* d_out, int out_size, void* d_ws, size_t ws_size,
                              hipStream_t stream) {
    (void)in_sizes; (void)n_in; (void)out_size; (void)d_ws; (void)ws_size;
    const float* mfcc0 = (const float*)d_in[0];
    const float* mfcc1 = (const float*)d_in[1];
    const float* mfcc2 = (const float*)d_in[2];
    const float* len0  = (const float*)d_in[3];
    const float* W_ih  = (const float*)d_in[4];
    const float* W_hh  = (const float*)d_in[5];
    const float* b_ih  = (const float*)d_in[6];
    const float* b_hh  = (const float*)d_in[7];
    const float* W_out = (const float*)d_in[8];
    const float* b_out = (const float*)d_in[9];
    float* out = (float*)d_out;

    gru_fused_kernel<<<NWG, NTHREADS, 0, stream>>>(
        mfcc0, mfcc1, mfcc2, len0, W_ih, W_hh, b_ih, b_hh, W_out, b_out, out);
}
